// RetainAttention_83485574300220
// MI455X (gfx1250) — matrix operand order repaired and checked by reading, not compiled, not run on hardware
//
#include <hip/hip_runtime.h>
#include <hip/hip_bf16.h>

// Problem constants
#define B_  32
#define S_  2048
#define H_  512
#define E_  512
#define M_  (B_*S_)        // 65536 rows (b,s flattened)
#define CH_ 16             // scan chunks along S
#define SL_ (S_/CH_)       // 128 steps per chunk
#define TPC_ (SL_/16)      // 8 GEMM m-tiles per chunk

typedef __attribute__((ext_vector_type(16))) __bf16 v16bf;
typedef __attribute__((ext_vector_type(8)))  __bf16 v8bf;
typedef __attribute__((ext_vector_type(8)))  float  v8f;

// ---------------------------------------------------------------------------
// Kernel W: pre-convert Wb [E,H] f32 -> bf16 once (512 KB, lives in L2).
// ---------------------------------------------------------------------------
__global__ __launch_bounds__(256)
void k_cvt_wb(const float* __restrict__ Wb, __bf16* __restrict__ wbbf) {
  const int gid  = blockIdx.x * 256 + threadIdx.x;   // E_*H_/8 threads
  const size_t base = (size_t)gid * 8;
  float4 a = *(const float4*)(Wb + base);
  float4 b = *(const float4*)(Wb + base + 4);
  v8bf o;
  o[0]=(__bf16)a.x; o[1]=(__bf16)a.y; o[2]=(__bf16)a.z; o[3]=(__bf16)a.w;
  o[4]=(__bf16)b.x; o[5]=(__bf16)b.y; o[6]=(__bf16)b.z; o[7]=(__bf16)b.w;
  *(v8bf*)(wbbf + base) = o;
}

// ---------------------------------------------------------------------------
// Kernel A: w[m] = exp(dot(alpha[m,:], Wa) + ba).  One wave per row.
// ---------------------------------------------------------------------------
__global__ __launch_bounds__(256)
void k_rowdot_exp(const float* __restrict__ alpha,
                  const float* __restrict__ Wa,
                  const float* __restrict__ ba,
                  float* __restrict__ wout) {
  const int lane = threadIdx.x & 31;
  const int wave = threadIdx.x >> 5;
  const int m    = blockIdx.x * 8 + wave;
  const float4* ap = (const float4*)(alpha + (size_t)m * H_ + lane * 16);
  const float4* wp = (const float4*)(Wa + lane * 16);
  float s = 0.f;
#pragma unroll
  for (int i = 0; i < 4; ++i) {
    float4 a = ap[i], w = wp[i];
    s += a.x * w.x + a.y * w.y + a.z * w.z + a.w * w.w;
  }
#pragma unroll
  for (int off = 16; off >= 1; off >>= 1)
    s += __shfl_xor(s, off, 32);
  if (lane == 0) wout[m] = expf(s + ba[0]);
}

// ---------------------------------------------------------------------------
// Kernel B: wv[m,e] = w[m] * tanh( (X @ Wb^T)[m,e] + bb[e] ) * embed[m,e]
// bf16 WMMA (f32 accumulate). One wave = 16 rows x 256 cols.
// Single base pointers + immediate offsets; B-fragment double buffered.
// Also emits deterministic per-tile column sums (for the scan) to ppart.
// ---------------------------------------------------------------------------
__global__ __launch_bounds__(32)
void k_gemm_tanh_scale(const float* __restrict__ X,     // data_beta [M,H]
                       const __bf16* __restrict__ Wbbf, // [E,H] bf16
                       const float* __restrict__ bb,    // [E]
                       const float* __restrict__ emb,   // [M,E]
                       const float* __restrict__ wvec,  // w [M]
                       float* __restrict__ out,         // [M,E] (wv)
                       float* __restrict__ ppart) {     // [B,CH,TPC,E]
  const int lane = threadIdx.x & 31;
  const int half = lane >> 4;    // 0: lanes 0-15, 1: lanes 16-31
  const int l16  = lane & 15;
  const int mt   = blockIdx.x * 16;
  const int e0   = blockIdx.y * 256;

  v8f zero = {0.f,0.f,0.f,0.f,0.f,0.f,0.f,0.f};
  v8f acc[16];
#pragma unroll
  for (int i = 0; i < 16; ++i) acc[i] = zero;

  // Loop-invariant base pointers; per-kt / per-nt deltas are compile-time
  // constants that fold into the 24-bit instruction offsets.
  const float*  arow = X + (size_t)(mt + l16) * H_ + half * 8;         // A row
  const __bf16* bcol = Wbbf + (size_t)(e0 + l16) * H_ + half * 8;      // B col, nt=0

  for (int kt = 0; kt < H_ / 32; ++kt) {
    const int k0 = kt * 32;
    // ---- A fragment (f32 -> bf16, amortized over 16 B-tiles).
    // 16-bit A 16x32 layout:
    //  lane<16 : VGPR0-3 -> K=k0..k0+7,    VGPR4-7 -> K=k0+16..k0+23
    //  lane>=16: VGPR0-3 -> K=k0+8..k0+15, VGPR4-7 -> K=k0+24..k0+31
    const float* abase = arow + k0;
    float4 a0 = *(const float4*)(abase);
    float4 a1 = *(const float4*)(abase + 4);
    float4 a2 = *(const float4*)(abase + 16);
    float4 a3 = *(const float4*)(abase + 20);
    v16bf af;
    af[0]=(__bf16)a0.x; af[1]=(__bf16)a0.y; af[2]=(__bf16)a0.z; af[3]=(__bf16)a0.w;
    af[4]=(__bf16)a1.x; af[5]=(__bf16)a1.y; af[6]=(__bf16)a1.z; af[7]=(__bf16)a1.w;
    af[8]=(__bf16)a2.x; af[9]=(__bf16)a2.y; af[10]=(__bf16)a2.z; af[11]=(__bf16)a2.w;
    af[12]=(__bf16)a3.x; af[13]=(__bf16)a3.y; af[14]=(__bf16)a3.z; af[15]=(__bf16)a3.w;

    // ---- B fragments: pre-converted bf16, double-buffered b128 loads.
    // B 32x16 layout: lane owns column N=l16; K = k0+8*(lane/16)+0..7, k0+16+8*(lane/16)+0..7 (same as A).
    // Per-nt delta: 16 columns * H_ elements = constant byte offset.
    const __bf16* bk = bcol + k0;
    v8bf lo[2], hi[2];
    lo[0] = *(const v8bf*)(bk);
    hi[0] = *(const v8bf*)(bk + 16);
#pragma unroll
    for (int nt = 0; nt < 16; ++nt) {
      const int cur = nt & 1, nxt = cur ^ 1;
      if (nt < 15) {  // prefetch next B tile before consuming current
        lo[nxt] = *(const v8bf*)(bk + (nt + 1) * 16 * H_);
        hi[nxt] = *(const v8bf*)(bk + (nt + 1) * 16 * H_ + 16);
      }
      v16bf bf = __builtin_shufflevector(lo[cur], hi[cur],
          0,1,2,3,4,5,6,7,8,9,10,11,12,13,14,15);
      acc[nt] = __builtin_amdgcn_wmma_f32_16x16x32_bf16(
          false, af, false, bf, (short)0, acc[nt], false, false);
    }
  }

  // ---- Epilogue. C/D layout: VGPR r holds M = mt + r + 8*half, N = l16.
  float wr8[8];
#pragma unroll
  for (int r = 0; r < 8; ++r) wr8[r] = wvec[mt + r + half * 8];

  // tile position for partial-sum output
  const int b    = mt / S_;
  const int srow = mt % S_;
  const int c    = srow / SL_;
  const int tic  = (srow % SL_) / 16;   // tile-in-chunk, 0..7

#pragma unroll
  for (int nt = 0; nt < 16; ++nt) {
    const int eg = e0 + nt * 16 + l16;
    const float bias = bb[eg];
    float cs = 0.f;
#pragma unroll
    for (int r = 0; r < 8; ++r) {
      const int m = mt + r + half * 8;
      const float t = tanhf(acc[nt][r] + bias);
      const float val = t * emb[(size_t)m * E_ + eg] * wr8[r];
      out[(size_t)m * E_ + eg] = val;
      cs += val;
    }
    // full 16-row column sum = this half + other half
    cs += __shfl_xor(cs, 16, 32);
    if (half == 0)
      ppart[((((size_t)b * CH_ + c) * TPC_) + tic) * E_ + eg] = cs;
  }
}

// ---------------------------------------------------------------------------
// Kernel C': chunk sums of w (scalar per (b,chunk)); tiny.
// ---------------------------------------------------------------------------
__global__ __launch_bounds__(128)
void k_w_chunk_sums(const float* __restrict__ warr, float* __restrict__ psumw) {
  __shared__ float sm[SL_];
  const int b = blockIdx.x / CH_;
  const int c = blockIdx.x % CH_;
  sm[threadIdx.x] = warr[(size_t)b * S_ + c * SL_ + threadIdx.x];
  __syncthreads();
  if (threadIdx.x == 0) {
    float s = 0.f;
    for (int i = 0; i < SL_; ++i) s += sm[i];
    psumw[b * CH_ + c] = s;
  }
}

// ---------------------------------------------------------------------------
// Kernel D: fold tile partials into chunk sums + exclusive scan over chunks.
// ---------------------------------------------------------------------------
__global__ __launch_bounds__(256)
void k_scan_chunks(const float* __restrict__ ppart,  // [B,CH,TPC,E]
                   float* __restrict__ psum,         // [B,CH,E] exclusive out
                   float* __restrict__ psumw) {      // [B,CH] in-place scan
  const int gid = blockIdx.x * blockDim.x + threadIdx.x;  // B_*E_ threads
  const int b = gid / E_;
  const int e = gid % E_;
  float run = 0.f;
#pragma unroll
  for (int c = 0; c < CH_; ++c) {
    float tot = 0.f;
#pragma unroll
    for (int t = 0; t < TPC_; ++t)
      tot += ppart[((((size_t)b * CH_ + c) * TPC_) + t) * E_ + e];
    psum[((size_t)b * CH_ + c) * E_ + e] = run;
    run += tot;
  }
  if (gid < B_) {
    float r2 = 0.f;
    for (int c = 0; c < CH_; ++c) {
      const int idx = gid * CH_ + c;
      const float t = psumw[idx];
      psumw[idx] = r2;
      r2 += t;
    }
  }
}

// ---------------------------------------------------------------------------
// Kernel E: apply prefix within chunk, divide by running denominator,
// overwrite d_out in place.
// ---------------------------------------------------------------------------
__global__ __launch_bounds__(512)
void k_scan_apply(float* __restrict__ out,           // in: wv, out: result
                  const float* __restrict__ warr,    // [M]
                  const float* __restrict__ psum,    // [B,CH,E] exclusive
                  const float* __restrict__ psumw) { // [B,CH]  exclusive
  __shared__ float wsm[SL_];
  const int b = blockIdx.x / CH_;
  const int c = blockIdx.x % CH_;
  const int e = threadIdx.x;
  const size_t rowbase = (size_t)b * S_ + c * SL_;
  if (threadIdx.x < SL_) wsm[threadIdx.x] = warr[rowbase + threadIdx.x];
  __syncthreads();
  float acc = psum[((size_t)b * CH_ + c) * E_ + e];
  float den = psumw[b * CH_ + c];
#pragma unroll 4
  for (int s = 0; s < SL_; ++s) {
    const size_t idx = (rowbase + s) * E_ + e;
    acc += out[idx];
    den += wsm[s];
    out[idx] = acc / (den + 1e-10f);
  }
}

// ---------------------------------------------------------------------------
extern "C" void kernel_launch(void* const* d_in, const int* in_sizes, int n_in,
                              void* d_out, int out_size, void* d_ws, size_t ws_size,
                              hipStream_t stream) {
  (void)in_sizes; (void)n_in; (void)out_size; (void)ws_size;
  const float* alpha = (const float*)d_in[0];   // [B,S,H]
  const float* beta  = (const float*)d_in[1];   // [B,S,H]
  const float* embed = (const float*)d_in[2];   // [B,S,E]
  const float* Wb    = (const float*)d_in[3];   // [E,H]
  const float* bb    = (const float*)d_in[4];   // [E]
  const float* Wa    = (const float*)d_in[5];   // [1,H]
  const float* ba    = (const float*)d_in[6];   // [1]
  float* out = (float*)d_out;

  // Workspace layout (floats):
  float* ws    = (float*)d_ws;
  float*  warr  = ws;                                   // M_            (256 KB)
  __bf16* wbbf  = (__bf16*)(ws + M_);                   // E_*H_ bf16    (512 KB)
  float*  ppart = ws + M_ + (E_ * H_) / 2;              // B*CH*TPC*E    (8 MB)
  float*  psum  = ppart + (size_t)B_ * CH_ * TPC_ * E_; // B*CH*E        (1 MB)
  float*  psumw = psum + (size_t)B_ * CH_ * E_;         // B*CH          (2 KB)

  k_cvt_wb<<<(E_ * H_ / 8) / 256, 256, 0, stream>>>(Wb, wbbf);
  k_rowdot_exp<<<M_ / 8, 256, 0, stream>>>(alpha, Wa, ba, warr);

  dim3 g2(M_ / 16, E_ / 256);
  k_gemm_tanh_scale<<<g2, 32, 0, stream>>>(beta, wbbf, bb, embed, warr, out, ppart);

  k_w_chunk_sums<<<B_ * CH_, 128, 0, stream>>>(warr, psumw);
  k_scan_chunks<<<(B_ * E_) / 256, 256, 0, stream>>>(ppart, psum, psumw);
  k_scan_apply<<<B_ * CH_, 512, 0, stream>>>(out, warr, psum, psumw);
}
